// DiffusionDynamicInput_79336635892253
// MI455X (gfx1250) — matrix operand order repaired and checked by reading, not compiled, not run on hardware
//
#include <hip/hip_runtime.h>
#include <hip/hip_bf16.h>
#include <math.h>

// DiffusionDynamicInput: dynamic per-(b,band) 3x3 conv fused over bands.
// B=8 N=13 H=W=256 D_EMB=1024 D_OUT=128 K=3
//
// Main op per batch b: GEMM [d=128] x [k] @ [k] x [pix=65536], k = 13 bands * 9 taps.
// 15.7 GFLOP, ~300 MB HBM (268 MB f32 output dominates) => ~13 us floor @ 23.3 TB/s.
// v_wmma_f32_16x16x32_bf16 with K-ordering k' = tap*16 + band (padded to 160):
//  - x pre-packed channel-last bf16 with zero halo: xpad[b][y][x][16] -> each
//    B fragment is exactly two global_load_b128; all 5 fragments preloaded and
//    pinned above the K-loop with sched_barrier.
//  - weights staged in LDS with 84-u32 row stride (bank-conflict-free b128 reads),
//    A fragments prefetched at distance 2 (3 rotating buffers, sched_barrier-pinned)
//    so ds loads overlap WMMA execution instead of wait-to-zero lockstep.
//  - bias pre-loaded into the WMMA accumulator.

#define BATCH 8
#define NB    13
#define HT    256
#define WD    256
#define DEMB  1024
#define DHID  512           // 4*D_OUT
#define DOUT  128
#define K2    9
#define KCP   160           // padded K': tap*16 + band, 5 k-blocks of 32

#define WSTR  84            // weight row stride in u32 (>=80, conflict-free in LDS)
#define WROW  (WSTR*2)      // 168 bf16 per d-row
#define WGTN  (DOUT*WSTR)   // 10752 u32 per batch

#define PADW  258           // 256 + 1 halo each side
#define NBP   16            // bands padded to 16 (13..15 zero)

typedef __attribute__((ext_vector_type(16))) __bf16 v16bf;
typedef __attribute__((ext_vector_type(8)))  float  v8f;

union Frag {
    v16bf        v;
    unsigned int u[8];
    uint4        q[2];
};

union Pack16 {
    __bf16 h[16];
    uint4  q[2];
};

// ---------------- stage 0: pack x into zero-padded channel-last bf16 ----------------
// xpad[b][y][x][band0..15], (y,x) in 258x258 with 1-px zero halo, bands 13..15 zero
__global__ void pack_kernel(const float* __restrict__ x, __bf16* __restrict__ xpad) {
    int b = blockIdx.y;                // 0..7
    int y = blockIdx.x;                // 0..257
    bool rowok = (y >= 1) && (y <= HT);
    for (int xc = threadIdx.x; xc < PADW; xc += 256) {
        Pack16 t;
        bool ok = rowok && (xc >= 1) && (xc <= WD);
#pragma unroll
        for (int n = 0; n < NB; ++n) {
            float v = 0.0f;
            if (ok) v = x[((size_t)(b * NB + n) * HT + (y - 1)) * WD + (xc - 1)];
            t.h[n] = (__bf16)v;
        }
        t.h[13] = (__bf16)0.0f; t.h[14] = (__bf16)0.0f; t.h[15] = (__bf16)0.0f;
        uint4* dst = (uint4*)(xpad + ((size_t)(b * PADW + y) * PADW + xc) * NBP);
        dst[0] = t.q[0];
        dst[1] = t.q[1];
    }
}

// zero padded bf16 weight buffer (bands 13..15, k' in [144,168) must stay 0)
__global__ void zero_wgt_kernel(unsigned int* __restrict__ wgt32) {
    int i = blockIdx.x * blockDim.x + threadIdx.x;   // exactly 8*10752 = 86016 threads
    wgt32[i] = 0u;
}

// ---------------- stage 1: weight/bias generation (tiny GEMMs) ----------------
// 8 context rows per block so w1 streams from L2 13x instead of 104x.
__global__ void hid_kernel(const float* __restrict__ wv,
                           const float* __restrict__ temb,
                           const float* __restrict__ w1,
                           const float* __restrict__ b1,
                           float* __restrict__ hid) {
    __shared__ float sctx[8][DEMB];    // 32 KB
    int bn0 = blockIdx.x * 8;
    int tid = threadIdx.x;             // blockDim = 512
    for (int i = tid; i < 8 * DEMB; i += 512) {
        int g = i >> 10, e = i & (DEMB - 1);
        int bn = bn0 + g, b = bn / NB;
        sctx[g][e] = wv[bn * DEMB + e] + temb[b * DEMB + e];
    }
    __syncthreads();
    int   o = tid;
    float a[8];
#pragma unroll
    for (int g = 0; g < 8; ++g) a[g] = b1[o];
#pragma unroll 4
    for (int e = 0; e < DEMB; ++e) {
        float wval = w1[e * DHID + o];
#pragma unroll
        for (int g = 0; g < 8; ++g) a[g] = fmaf(sctx[g][e], wval, a[g]);
    }
#pragma unroll
    for (int g = 0; g < 8; ++g) {
        float s = a[g];
        s = s / (1.0f + __expf(-s));   // SiLU
        hid[(bn0 + g) * DHID + o] = s;
    }
}

// weights -> bf16 at wgt[b][d][k' = tap*16 + band], row stride WROW
__global__ void wgt_kernel(const float* __restrict__ hid,
                           const float* __restrict__ w2,
                           const float* __restrict__ b2,
                           __bf16* __restrict__ wgt) {
    __shared__ float shid[8][DHID];    // 16 KB
    int bn0 = blockIdx.x * 8;
    int tid = threadIdx.x;             // blockDim = 384
    for (int i = tid; i < 8 * DHID; i += 384) {
        int g = i >> 9, e = i & (DHID - 1);
        shid[g][e] = hid[(bn0 + g) * DHID + e];
    }
    __syncthreads();
    for (int j = tid; j < DOUT * K2; j += 384) {
        float a[8];
#pragma unroll
        for (int g = 0; g < 8; ++g) a[g] = b2[j];
#pragma unroll 4
        for (int e = 0; e < DHID; ++e) {
            float wval = w2[e * (DOUT * K2) + j];
#pragma unroll
            for (int g = 0; g < 8; ++g) a[g] = fmaf(shid[g][e], wval, a[g]);
        }
        int d = j / K2;
        int p = j - d * K2;            // tap index (kh*3+kw)
#pragma unroll
        for (int g = 0; g < 8; ++g) {
            int bn = bn0 + g, b = bn / NB, n = bn - b * NB;
            wgt[(size_t)(b * DOUT + d) * WROW + (p * 16 + n)] = (__bf16)a[g];
        }
    }
}

// bias[b][d] = (sum_n ctx[b,n]) @ wb + NB*bb
__global__ void bias_kernel(const float* __restrict__ wv,
                            const float* __restrict__ temb,
                            const float* __restrict__ wb,
                            const float* __restrict__ bb,
                            float* __restrict__ biasv) {
    __shared__ float ssum[DEMB];
    int b   = blockIdx.x;
    int tid = threadIdx.x;             // blockDim = 128
    for (int e = tid; e < DEMB; e += 128) {
        float s = (float)NB * temb[b * DEMB + e];
        for (int n = 0; n < NB; ++n) s += wv[(b * NB + n) * DEMB + e];
        ssum[e] = s;
    }
    __syncthreads();
    int   d = tid;
    float s = (float)NB * bb[d];
#pragma unroll 8
    for (int e = 0; e < DEMB; ++e) s = fmaf(ssum[e], wb[e * DOUT + d], s);
    biasv[b * DOUT + d] = s;
}

// ---------------- stage 2: main WMMA dynamic-conv GEMM ----------------
// A (16 d x 32 k') = generated weights (LDS, bf16, stride-84 rows)
// B (32 k' x 16 pixels): per k-block one tap per lane-half -> 2x global_load_b128
// C (16 d x 16 pix, f32), bias pre-loaded into the accumulator.
__global__ __launch_bounds__(256)
void dynconv_kernel(const __bf16* __restrict__ xpad,
                    const unsigned int* __restrict__ wgt32,
                    const float* __restrict__ biasv,
                    float* __restrict__ out) {
    __shared__ unsigned int lds_w[WGTN];   // 10752 u32 = 42 KB, [d][stride 84]
    __shared__ float lds_bias[DOUT];

    int b   = blockIdx.y;
    int tid = threadIdx.x;

    // flat uint4 memcpy of this batch's padded weights
    {
        const uint4* gsrc = (const uint4*)(wgt32 + (size_t)b * WGTN);
        uint4* ldst = (uint4*)lds_w;
        for (int i = tid; i < WGTN / 4; i += 256) ldst[i] = gsrc[i];
    }
    if (tid < DOUT) lds_bias[tid] = biasv[b * DOUT + tid];
    __syncthreads();

    int wave = tid >> 5;
    int lane = tid & 31;
    int lh   = lane >> 4;                       // lane half
    int ln   = lane & 15;
    int m0   = blockIdx.x * 128 + wave * 16;    // pixel tile base (half an h row per block)
    int pix  = m0 + ln;
    int h    = pix >> 8;
    int w    = pix & (WD - 1);

    // center-tap address of this lane's pixel (channel-last padded space)
    const __bf16* xb = xpad + ((size_t)(b * PADW + (h + 1)) * PADW + (w + 1)) * NBP;

    // tap offsets (dy*258+dx) per k-block for lane-half 0 / 1; tap p = kb*2+lh.
    // p=9 (invalid, kb=4,lh=1) aliases tap 8 -- its weights are zero.
    const int OFF0[5] = {-259, -257,  0, 257, 259};
    const int OFF1[5] = {-258,   -1,  1, 258, 259};

    // ---- preload ALL 5 B fragments (10 global_load_b128) ----
    Frag bfr[5];
#pragma unroll
    for (int kb = 0; kb < 5; ++kb) {
        const uint4* src = (const uint4*)(xb + (lh ? OFF1[kb] : OFF0[kb]) * NBP);
        bfr[kb].q[0] = src[0];
        bfr[kb].q[1] = src[1];
    }

    // bias -> accumulator init (C layout: d = t*16 + lh*8 + r, pix col = ln)
    v8f acc[8];
#pragma unroll
    for (int t = 0; t < 8; ++t)
#pragma unroll
        for (int r = 0; r < 8; ++r)
            acc[t][r] = lds_bias[t * 16 + lh * 8 + r];

    // A-fragment loader: contiguous K layout (lanes0-15: K0-15; lanes16-31: K16-31)
    auto loadA = [&](Frag& dst, int i) {
        int kb = i >> 3, t = i & 7;
        int base = (t * 16 + ln) * WSTR + kb * 16 + lh * 8;
#pragma unroll
        for (int v = 0; v < 8; ++v) dst.u[v] = lds_w[base + v];      // K 0..15 (+16lh)
    };

    // pin the B preload + acc init above the K-loop
    __builtin_amdgcn_sched_barrier(0);

    // ---- K-loop: 40 WMMAs, A prefetched at distance 2 (3 rotating buffers) ----
    Frag aw[3];
    loadA(aw[0], 0);
    loadA(aw[1], 1);
#pragma unroll
    for (int i = 0; i < 40; ++i) {
        if (i + 2 < 40) loadA(aw[(i + 2) % 3], i + 2);
        __builtin_amdgcn_sched_barrier(0);   // keep prefetch above this WMMA
        acc[i & 7] = __builtin_amdgcn_wmma_f32_16x16x32_bf16(
            false, aw[i % 3].v, false, bfr[i >> 3].v, (short)0, acc[i & 7], false, false);
    }

    // ---- store (bias already in acc) ----
#pragma unroll
    for (int t = 0; t < 8; ++t)
#pragma unroll
        for (int r = 0; r < 8; ++r) {
            int d = t * 16 + lh * 8 + r;
            out[((size_t)b * DOUT + d) * (HT * WD) + m0 + ln] = acc[t][r];
        }
}

// ---------------- launch ----------------

extern "C" void kernel_launch(void* const* d_in, const int* in_sizes, int n_in,
                              void* d_out, int out_size, void* d_ws, size_t ws_size,
                              hipStream_t stream) {
    const float* x    = (const float*)d_in[0];
    const float* temb = (const float*)d_in[1];
    const float* wv   = (const float*)d_in[2];
    const float* w1   = (const float*)d_in[3];
    const float* b1   = (const float*)d_in[4];
    const float* w2   = (const float*)d_in[5];
    const float* b2   = (const float*)d_in[6];
    const float* wb   = (const float*)d_in[7];
    const float* bb   = (const float*)d_in[8];
    float* out = (float*)d_out;

    // workspace layout
    char* ws = (char*)d_ws;
    float*  hid   = (float*)ws;                      // 104*512*4        = 212992 B
    __bf16* wgt   = (__bf16*)(ws + 212992);          // 8*10752*4        = 344064 B
    float*  biasv = (float*)(ws + 212992 + 344064);  // 8*128*4          =   4096 B
    __bf16* xpad  = (__bf16*)(ws + 212992 + 344064 + 4096); // 8*258*258*16*2 = 17040384 B
    unsigned int* wgt32 = (unsigned int*)wgt;

    hipLaunchKernelGGL(pack_kernel, dim3(PADW, BATCH), dim3(256), 0, stream, x, xpad);
    hipLaunchKernelGGL(zero_wgt_kernel, dim3(336), dim3(256), 0, stream, wgt32);
    hipLaunchKernelGGL(hid_kernel,  dim3(BATCH * NB / 8), dim3(512), 0, stream,
                       wv, temb, w1, b1, hid);
    hipLaunchKernelGGL(wgt_kernel,  dim3(BATCH * NB / 8), dim3(384), 0, stream,
                       hid, w2, b2, wgt);
    hipLaunchKernelGGL(bias_kernel, dim3(BATCH), dim3(128), 0, stream,
                       wv, temb, wb, bb, biasv);
    hipLaunchKernelGGL(dynconv_kernel, dim3(HT * WD / 128, BATCH), dim3(256), 0, stream,
                       xpad, wgt32, biasv, out);
}
